// InteractionNetwork_62947040690377
// MI455X (gfx1250) — matrix operand order repaired and checked by reading, not compiled, not run on hardware
//
#include <hip/hip_runtime.h>
#include <hip/hip_bf16.h>

// ---------------------------------------------------------------------------
// InteractionNetwork on MI455X (gfx1250): fused WMMA implementation.
//   N=50000 nodes, E=400000 edges, FN=FE=128, W=256.
//   ~190 GFLOP of GEMM, ~0.5 GB HBM traffic -> must run on v_wmma_f32_16x16x32_f16.
// ---------------------------------------------------------------------------

#define NN 50000
#define EE 400000
#define BM 32            // rows (edges/nodes) per workgroup
#define NTHREADS 256     // 8 waves of 32

typedef __attribute__((ext_vector_type(16))) _Float16 v16h;
typedef __attribute__((ext_vector_type(8)))  _Float16 v8h;
typedef __attribute__((ext_vector_type(8)))  float    v8f;

// ---------------------------------------------------------------------------
// One 16x16 output tile, K = KI*32, A from LDS (f16, row-major, stride lda),
// B from global f16 weights stored N-major: bT[n*ldb + k].
//
// CDNA5 16-bit A fragment (16x32): lane l (m=l&15, g=l>>4) holds
//   halves 0..7  = A[m][kk*32 + g*8 .. +7]        (contiguous, one b128)
//   halves 8..15 = A[m][kk*32 + 16 + g*8 .. +7]   (contiguous, one b128)
// CDNA5 16-bit B fragment (32x16): lane l (n=l&15, g=l>>4) holds
//   halves 0..7 = bT[n][kk*32 + g*8 .. +7], halves 8..15 = bT[n][kk*32 + 16 + g*8 .. +7]
// C/D f32: element u of lane l is D[mrow + u + g*8][ncol + n].
// ---------------------------------------------------------------------------
template <int KI>
__device__ __attribute__((always_inline)) inline
v8f gemm_tile(const _Float16* __restrict__ a, int lda,
              const _Float16* __restrict__ bT, int ldb,
              int mrow, int ncol, v8f acc) {
  const int lane = threadIdx.x & 31;
  const int lo = lane & 15;
  const int g = lane >> 4;
  const _Float16* arow = a + (size_t)(mrow + lo) * lda + g * 8;
  const _Float16* brow = bT + (size_t)(ncol + lo) * ldb + g * 8;
#pragma unroll
  for (int kk = 0; kk < KI; ++kk) {
    union { v16h v; v8h h[2]; } A;
    A.h[0] = *(const v8h*)(arow + kk * 32);
    A.h[1] = *(const v8h*)(arow + kk * 32 + 16);
    union { v16h v; v8h h[2]; } Bu;
    Bu.h[0] = *(const v8h*)(brow + kk * 32);
    Bu.h[1] = *(const v8h*)(brow + kk * 32 + 16);
    v16h B = Bu.v;
    acc = __builtin_amdgcn_wmma_f32_16x16x32_f16(false, A.v, false, B,
                                                 (short)0, acc, false, false);
  }
  return acc;
}

// ---------------------------------------------------------------------------
// Prep kernels
// ---------------------------------------------------------------------------
__global__ void k_transpose_w(const float* __restrict__ src,
                              _Float16* __restrict__ dst, int K, int Nn) {
  int i = blockIdx.x * blockDim.x + threadIdx.x;
  if (i < K * Nn) {
    int k = i / Nn, n = i % Nn;
    dst[(size_t)n * K + k] = (_Float16)src[i];
  }
}

__global__ void k_zero(float* __restrict__ p, long n) {
  long i = (long)blockIdx.x * blockDim.x + threadIdx.x;
  if (i < n) p[i] = 0.0f;
}

// ---------------------------------------------------------------------------
// Edge kernel: e_new = e@W_l + relu-FNN(LN(concat[e, v[row], v[col]]))
// plus float-atomic scatter of e_new into aggr[col] and counts[col].
// ---------------------------------------------------------------------------
__global__ __launch_bounds__(NTHREADS) void k_edge(
    const float* __restrict__ v, const float* __restrict__ e,
    const int* __restrict__ eidx,
    const float* __restrict__ lng, const float* __restrict__ lnb,
    const _Float16* __restrict__ w0T, const float* __restrict__ b0,
    const _Float16* __restrict__ w1T, const float* __restrict__ b1,
    const _Float16* __restrict__ w2T, const float* __restrict__ b2,
    const _Float16* __restrict__ wlT,
    float* __restrict__ eout, float* __restrict__ aggr,
    float* __restrict__ counts) {
  // LDS: 25088 + 16896 + 8704 + 128 = ~50.8 KB
  __shared__ __align__(32) _Float16 s_x[BM * 392];   // LN(concat) f16; reused as h2
  __shared__ __align__(32) _Float16 s_h1[BM * 264];  // hidden 1
  __shared__ __align__(32) _Float16 s_eh[BM * 136];  // raw e (residual A)
  __shared__ int s_co[BM];

  const int tid = threadIdx.x;
  const long blk = blockIdx.x;

  // -------- Phase A: gather + concat + LayerNorm(384) --------
  {
    const int r = tid >> 3, q = tid & 7;      // 8 threads per row, 48 cols each
    const long edge = blk * BM + r;           // E % BM == 0: always valid
    const int ro = eidx[edge];
    const int co = eidx[EE + edge];
    if (q == 0) {
      s_co[r] = co;
      atomicAdd(&counts[co], 1.0f);
    }
    const float* pe = e + edge * 128;
    const float* pr = v + (long)ro * 128;
    const float* pc = v + (long)co * 128;
    float4 d[12];
    float s = 0.f, ss = 0.f;
#pragma unroll
    for (int j = 0; j < 12; ++j) {
      const int c = (q * 12 + j) * 4;  // float4 never crosses a 128-boundary
      const float* src = (c < 128) ? (pe + c)
                                   : ((c < 256) ? (pr + c - 128) : (pc + c - 256));
      float4 t = *(const float4*)src;
      d[j] = t;
      s += t.x + t.y + t.z + t.w;
      ss += t.x * t.x + t.y * t.y + t.z * t.z + t.w * t.w;
    }
    s  += __shfl_xor(s, 1, 32);  s  += __shfl_xor(s, 2, 32);  s  += __shfl_xor(s, 4, 32);
    ss += __shfl_xor(ss, 1, 32); ss += __shfl_xor(ss, 2, 32); ss += __shfl_xor(ss, 4, 32);
    const float mean = s * (1.f / 384.f);
    const float rstd = rsqrtf(ss * (1.f / 384.f) - mean * mean + 1e-5f);
#pragma unroll
    for (int j = 0; j < 12; ++j) {
      const int c = (q * 12 + j) * 4;
      const float tv[4] = {d[j].x, d[j].y, d[j].z, d[j].w};
#pragma unroll
      for (int u = 0; u < 4; ++u) {
        const int cc = c + u;
        s_x[r * 392 + cc] = (_Float16)((tv[u] - mean) * rstd * lng[cc] + lnb[cc]);
        if (c < 128) s_eh[r * 136 + cc] = (_Float16)tv[u];  // raw e for residual
      }
    }
  }
  __syncthreads();

  const int w = tid >> 5;
  const int lane = tid & 31;
  const int nlo = lane & 15, g = lane >> 4;

  // -------- Layer 0: [32x384] x [384x256] + b0, relu -> s_h1 --------
  for (int t = w; t < 32; t += 8) {
    const int mt = (t >> 4) * 16, nt = (t & 15) * 16;
    const float bv = b0[nt + nlo];
    v8f acc = {bv, bv, bv, bv, bv, bv, bv, bv};
    acc = gemm_tile<12>(s_x, 392, w0T, 384, mt, nt, acc);
#pragma unroll
    for (int u = 0; u < 8; ++u) {
      const float x = acc[u];
      s_h1[(mt + u + g * 8) * 264 + nt + nlo] = (_Float16)(x > 0.f ? x : 0.f);
    }
  }
  __syncthreads();

  // -------- Layer 1: [32x256] x [256x256] + b1, relu -> h2 (aliases s_x) ----
  for (int t = w; t < 32; t += 8) {
    const int mt = (t >> 4) * 16, nt = (t & 15) * 16;
    const float bv = b1[nt + nlo];
    v8f acc = {bv, bv, bv, bv, bv, bv, bv, bv};
    acc = gemm_tile<8>(s_h1, 264, w1T, 256, mt, nt, acc);
#pragma unroll
    for (int u = 0; u < 8; ++u) {
      const float x = acc[u];
      s_x[(mt + u + g * 8) * 264 + nt + nlo] = (_Float16)(x > 0.f ? x : 0.f);
    }
  }
  __syncthreads();

  // -------- Layer 2 (+relu) then residual e@W_l; store + atomic scatter ----
  for (int t = w; t < 16; t += 8) {
    const int mt = (t >> 3) * 16, nt = (t & 7) * 16;
    const float bv = b2[nt + nlo];
    v8f acc = {bv, bv, bv, bv, bv, bv, bv, bv};
    acc = gemm_tile<8>(s_x, 264, w2T, 256, mt, nt, acc);
#pragma unroll
    for (int u = 0; u < 8; ++u) {  // relu BEFORE residual (fnn relus last layer)
      const float x = acc[u];
      acc[u] = x > 0.f ? x : 0.f;
    }
    acc = gemm_tile<4>(s_eh, 136, wlT, 128, mt, nt, acc);
#pragma unroll
    for (int u = 0; u < 8; ++u) {
      const int rl = mt + u + g * 8;
      const long edge = blk * BM + rl;
      const int cf = nt + nlo;
      const float val = acc[u];
      eout[edge * 128 + cf] = val;
      atomicAdd(&aggr[(long)s_co[rl] * 128 + cf], val);
    }
  }
}

// ---------------------------------------------------------------------------
// Node kernel: v_new = v@W_l + relu-FNN(LN(concat[aggr/count, v]))
// ---------------------------------------------------------------------------
__global__ __launch_bounds__(NTHREADS) void k_node(
    const float* __restrict__ v,
    const float* __restrict__ aggr, const float* __restrict__ counts,
    const float* __restrict__ lng, const float* __restrict__ lnb,
    const _Float16* __restrict__ w0T, const float* __restrict__ b0,
    const _Float16* __restrict__ w1T, const float* __restrict__ b1,
    const _Float16* __restrict__ w2T, const float* __restrict__ b2,
    const _Float16* __restrict__ wlT,
    float* __restrict__ vout) {
  __shared__ __align__(32) _Float16 s_x[BM * 264];   // LN(concat); reused as h2
  __shared__ __align__(32) _Float16 s_h1[BM * 264];
  __shared__ __align__(32) _Float16 s_vh[BM * 136];  // raw v (residual A)

  const int tid = threadIdx.x;
  const long blk = blockIdx.x;

  // -------- Phase A: mean-aggregate read + concat + LayerNorm(256) --------
  {
    const int r = tid >> 3, q = tid & 7;  // 8 threads per row, 32 cols each
    const long node = blk * BM + r;
    const bool valid = node < NN;
    float inv = 1.f;
    if (valid) inv = 1.f / fmaxf(counts[node], 1.0f);
    const float* pa = aggr + node * 128;
    const float* pv = v + node * 128;
    float4 d[8];
    float s = 0.f, ss = 0.f;
#pragma unroll
    for (int j = 0; j < 8; ++j) {
      const int c = (q * 8 + j) * 4;
      float4 t = make_float4(0.f, 0.f, 0.f, 0.f);
      if (valid) {
        if (c < 128) {
          t = *(const float4*)(pa + c);
          t.x *= inv; t.y *= inv; t.z *= inv; t.w *= inv;
        } else {
          t = *(const float4*)(pv + c - 128);
        }
      }
      d[j] = t;
      s += t.x + t.y + t.z + t.w;
      ss += t.x * t.x + t.y * t.y + t.z * t.z + t.w * t.w;
    }
    s  += __shfl_xor(s, 1, 32);  s  += __shfl_xor(s, 2, 32);  s  += __shfl_xor(s, 4, 32);
    ss += __shfl_xor(ss, 1, 32); ss += __shfl_xor(ss, 2, 32); ss += __shfl_xor(ss, 4, 32);
    const float mean = s * (1.f / 256.f);
    const float rstd = rsqrtf(ss * (1.f / 256.f) - mean * mean + 1e-5f);
#pragma unroll
    for (int j = 0; j < 8; ++j) {
      const int c = (q * 8 + j) * 4;
      const float tv[4] = {d[j].x, d[j].y, d[j].z, d[j].w};
#pragma unroll
      for (int u = 0; u < 4; ++u) {
        const int cc = c + u;
        s_x[r * 264 + cc] = (_Float16)((tv[u] - mean) * rstd * lng[cc] + lnb[cc]);
        if (c >= 128) s_vh[r * 136 + (cc - 128)] = (_Float16)tv[u];  // raw v
      }
    }
  }
  __syncthreads();

  const int w = tid >> 5;
  const int lane = tid & 31;
  const int nlo = lane & 15, g = lane >> 4;

  // -------- Layer 0 --------
  for (int t = w; t < 32; t += 8) {
    const int mt = (t >> 4) * 16, nt = (t & 15) * 16;
    const float bv = b0[nt + nlo];
    v8f acc = {bv, bv, bv, bv, bv, bv, bv, bv};
    acc = gemm_tile<8>(s_x, 264, w0T, 256, mt, nt, acc);
#pragma unroll
    for (int u = 0; u < 8; ++u) {
      const float x = acc[u];
      s_h1[(mt + u + g * 8) * 264 + nt + nlo] = (_Float16)(x > 0.f ? x : 0.f);
    }
  }
  __syncthreads();

  // -------- Layer 1 -> h2 (aliases s_x) --------
  for (int t = w; t < 32; t += 8) {
    const int mt = (t >> 4) * 16, nt = (t & 15) * 16;
    const float bv = b1[nt + nlo];
    v8f acc = {bv, bv, bv, bv, bv, bv, bv, bv};
    acc = gemm_tile<8>(s_h1, 264, w1T, 256, mt, nt, acc);
#pragma unroll
    for (int u = 0; u < 8; ++u) {
      const float x = acc[u];
      s_x[(mt + u + g * 8) * 264 + nt + nlo] = (_Float16)(x > 0.f ? x : 0.f);
    }
  }
  __syncthreads();

  // -------- Layer 2 (+relu) + residual v@W_l; masked store --------
  for (int t = w; t < 16; t += 8) {
    const int mt = (t >> 3) * 16, nt = (t & 7) * 16;
    const float bv = b2[nt + nlo];
    v8f acc = {bv, bv, bv, bv, bv, bv, bv, bv};
    acc = gemm_tile<8>(s_x, 264, w2T, 256, mt, nt, acc);
#pragma unroll
    for (int u = 0; u < 8; ++u) {
      const float x = acc[u];
      acc[u] = x > 0.f ? x : 0.f;
    }
    acc = gemm_tile<4>(s_vh, 136, wlT, 128, mt, nt, acc);
#pragma unroll
    for (int u = 0; u < 8; ++u) {
      const long node = blk * BM + mt + u + g * 8;
      if (node < NN) vout[node * 128 + nt + nlo] = acc[u];
    }
  }
}

// ---------------------------------------------------------------------------
extern "C" void kernel_launch(void* const* d_in, const int* in_sizes, int n_in,
                              void* d_out, int out_size, void* d_ws,
                              size_t ws_size, hipStream_t stream) {
  (void)in_sizes; (void)n_in; (void)out_size; (void)ws_size;
  const float* v   = (const float*)d_in[0];
  const float* e   = (const float*)d_in[1];
  const int*   idx = (const int*)d_in[2];
  const float* elng = (const float*)d_in[3];
  const float* elnb = (const float*)d_in[4];
  const float* ew0 = (const float*)d_in[5];
  const float* eb0 = (const float*)d_in[6];
  const float* ew1 = (const float*)d_in[7];
  const float* eb1 = (const float*)d_in[8];
  const float* ew2 = (const float*)d_in[9];
  const float* eb2 = (const float*)d_in[10];
  const float* el  = (const float*)d_in[11];
  const float* nlng = (const float*)d_in[12];
  const float* nlnb = (const float*)d_in[13];
  const float* nw0 = (const float*)d_in[14];
  const float* nb0 = (const float*)d_in[15];
  const float* nw1 = (const float*)d_in[16];
  const float* nb1 = (const float*)d_in[17];
  const float* nw2 = (const float*)d_in[18];
  const float* nb2 = (const float*)d_in[19];
  const float* nl  = (const float*)d_in[20];

  // Workspace: f16 transposed weights (~786 KB) + aggr (25.6 MB) + counts.
  _Float16* ew0T = (_Float16*)d_ws;
  _Float16* ew1T = ew0T + 384 * 256;
  _Float16* ew2T = ew1T + 256 * 256;
  _Float16* elT  = ew2T + 256 * 128;
  _Float16* nw0T = elT  + 128 * 128;
  _Float16* nw1T = nw0T + 256 * 256;
  _Float16* nw2T = nw1T + 256 * 256;
  _Float16* nlT  = nw2T + 256 * 128;
  float* aggr   = (float*)(nlT + 128 * 128);
  float* counts = aggr + (long)NN * 128;

  float* vout = (float*)d_out;              // v_new first in tuple order
  float* eout = vout + (long)NN * 128;      // then e_new

  // --- prep: weight transposes (fp32 -> f16, N-major) + zero accumulators ---
  k_transpose_w<<<(384 * 256 + 255) / 256, 256, 0, stream>>>(ew0, ew0T, 384, 256);
  k_transpose_w<<<(256 * 256 + 255) / 256, 256, 0, stream>>>(ew1, ew1T, 256, 256);
  k_transpose_w<<<(256 * 128 + 255) / 256, 256, 0, stream>>>(ew2, ew2T, 256, 128);
  k_transpose_w<<<(128 * 128 + 255) / 256, 256, 0, stream>>>(el,  elT,  128, 128);
  k_transpose_w<<<(256 * 256 + 255) / 256, 256, 0, stream>>>(nw0, nw0T, 256, 256);
  k_transpose_w<<<(256 * 256 + 255) / 256, 256, 0, stream>>>(nw1, nw1T, 256, 256);
  k_transpose_w<<<(256 * 128 + 255) / 256, 256, 0, stream>>>(nw2, nw2T, 256, 128);
  k_transpose_w<<<(128 * 128 + 255) / 256, 256, 0, stream>>>(nl,  nlT,  128, 128);
  const long zn = (long)NN * 128 + NN;  // aggr + counts (contiguous)
  k_zero<<<(int)((zn + 255) / 256), 256, 0, stream>>>(aggr, zn);

  // --- edge pass (writes e_new, scatters into aggr/counts) ---
  k_edge<<<EE / BM, NTHREADS, 0, stream>>>(v, e, idx, elng, elnb,
                                           ew0T, eb0, ew1T, eb1, ew2T, eb2, elT,
                                           eout, aggr, counts);

  // --- node pass (reads aggr/counts, writes v_new) ---
  k_node<<<(NN + BM - 1) / BM, NTHREADS, 0, stream>>>(v, aggr, counts,
                                                      nlng, nlnb,
                                                      nw0T, nb0, nw1T, nb1,
                                                      nw2T, nb2, nlT, vout);
}
